// QuantLinear_9457517986097
// MI455X (gfx1250) — compile-verified, matrix operand order checked by reading, not run on hardware
//
#include <hip/hip_runtime.h>

// ---------------------------------------------------------------------------
// QuantLinear on MI455X (gfx1250): out = X @ ((q - zp) * scale)^T + bias
// GEMM M=8192, N=11008, K=4096 via v_wmma_f32_16x16x32_f16 (f32 accum).
// Per-wave 64x64 tile (4x4 WMMA frags), block 128x256 (8 waves, 2Mx4N).
// 4-deep circular LDS pipeline fed by global_load_async_to_lds_b128 (GVS
// addressing: SGPR base + loop-invariant per-lane offset), partial
// s_wait_asynccnt <= 12 so two stages stay in flight across the k-loop.
// ---------------------------------------------------------------------------

typedef __attribute__((ext_vector_type(16))) _Float16 v16h;
typedef __attribute__((ext_vector_type(8)))  _Float16 v8h;
typedef __attribute__((ext_vector_type(8)))  float    v8f;
typedef __attribute__((ext_vector_type(4)))  float    v4f;

#define M_TOT 8192      // B*S = 4*2048
#define K_TOT 4096      // IN
#define N_TOT 11008     // OUT

#define BM 128
#define BN 256
#define KC 32              // one WMMA K-step per stage
#define NKC (K_TOT / KC)   // 128

#define LDS_PAD 8
#define LDA_H   (KC + LDS_PAD)            // 40 halfs = 80 B row stride (16B mult.)
#define ATILE_H (BM * LDA_H)              // 5120 halfs
#define BTILE_H (BN * LDA_H)              // 10240 halfs
#define STAGE_H (ATILE_H + BTILE_H)       // 15360 halfs = 30 KB / stage
#define NSTAGE  4                         // 120 KB of the 320 KB WGP LDS

// --- CDNA5 async global->LDS, GVS form: saddr base + 32-bit lane offset -----
__device__ __forceinline__ void cp_b128_s(unsigned loff, unsigned goff,
                                          const _Float16* sbase) {
  asm volatile("global_load_async_to_lds_b128 %0, %1, %2"
               :: "v"(loff), "v"(goff), "s"(sbase) : "memory");
}

// wait until at most 12 async loads outstanding (= 2 stages x 6 loads/wave)
__device__ __forceinline__ void wait_async_12() {
#if __has_builtin(__builtin_amdgcn_s_wait_asynccnt)
  __builtin_amdgcn_s_wait_asynccnt(12);
#else
  asm volatile("s_wait_asynccnt 0xc" ::: "memory");
#endif
}

// --- prepack: fp32 activations -> f16 ---------------------------------------
__global__ void k_cvt_input(const float* __restrict__ x, _Float16* __restrict__ y) {
  size_t i = ((size_t)blockIdx.x * blockDim.x + threadIdx.x) * 8;
  v4f a = *(const v4f*)(x + i);
  v4f b = *(const v4f*)(x + i + 4);
  v8h h;
  h[0] = (_Float16)a[0]; h[1] = (_Float16)a[1];
  h[2] = (_Float16)a[2]; h[3] = (_Float16)a[3];
  h[4] = (_Float16)b[0]; h[5] = (_Float16)b[1];
  h[6] = (_Float16)b[2]; h[7] = (_Float16)b[3];
  *(v8h*)(y + i) = h;
}

// --- prepack: int32 qweight -> f16 of (q - zp[o])  (exact: |q-zp| <= 255) ---
__global__ void k_cvt_weight(const int* __restrict__ q, const int* __restrict__ zp,
                             _Float16* __restrict__ y) {
  size_t i = ((size_t)blockIdx.x * blockDim.x + threadIdx.x) * 8;
  const int o = (int)(i >> 12);          // K_TOT = 4096 = 2^12
  const int z = zp[o];
  int4 a = *(const int4*)(q + i);
  int4 b = *(const int4*)(q + i + 4);
  v8h h;
  h[0] = (_Float16)(float)(a.x - z); h[1] = (_Float16)(float)(a.y - z);
  h[2] = (_Float16)(float)(a.z - z); h[3] = (_Float16)(float)(a.w - z);
  h[4] = (_Float16)(float)(b.x - z); h[5] = (_Float16)(float)(b.y - z);
  h[6] = (_Float16)(float)(b.z - z); h[7] = (_Float16)(float)(b.w - z);
  *(v8h*)(y + i) = h;
}

// --- main GEMM ---------------------------------------------------------------
union Frag { v16h h; uint4 q[2]; };

__global__ __launch_bounds__(256) void k_qgemm(
    const _Float16* __restrict__ Ah,   // [M_TOT, K_TOT] f16
    const _Float16* __restrict__ Wh,   // [N_TOT, K_TOT] f16 (q - zp)
    const float*    __restrict__ w_scale,
    const float*    __restrict__ bias,
    float*          __restrict__ out)  // [M_TOT, N_TOT]
{
  __shared__ __align__(16) _Float16 lds[NSTAGE * STAGE_H];   // 120 KB

  const int t     = threadIdx.x;
  const int mBase = blockIdx.y * BM;
  const int nBase = blockIdx.x * BN;

  // ---- loop-invariant loader offsets (bytes) ------------------------------
  // chunk idx = t + c*256 -> row = t/4 + 64c, seg = t&3 (16B chunk in row)
  const int rw = t >> 2, sg = t & 3;
  unsigned vOff[4], lA[2], lB[4];
#pragma unroll
  for (int c = 0; c < 4; ++c)
    vOff[c] = (unsigned)((rw + 64 * c) * (K_TOT * 2) + sg * 16); // global: row*8KB
#pragma unroll
  for (int c = 0; c < 2; ++c)
    lA[c] = (unsigned)((rw + 64 * c) * (LDA_H * 2) + sg * 16);
#pragma unroll
  for (int c = 0; c < 4; ++c)
    lB[c] = (unsigned)(ATILE_H * 2 + (rw + 64 * c) * (LDA_H * 2) + sg * 16);

  const unsigned ldsBase = (unsigned)(unsigned long long)(size_t)lds;
  const _Float16* Apanel = Ah + (size_t)mBase * K_TOT;
  const _Float16* Bpanel = Wh + (size_t)nBase * K_TOT;

  auto issue = [&](unsigned stageOff, int kc) {
    const _Float16* sA = Apanel + kc * KC;   // uniform -> SGPR base
    const _Float16* sB = Bpanel + kc * KC;
#pragma unroll
    for (int c = 0; c < 2; ++c) cp_b128_s(stageOff + lA[c], vOff[c], sA);
#pragma unroll
    for (int c = 0; c < 4; ++c) cp_b128_s(stageOff + lB[c], vOff[c], sB);
  };

  // ---- WMMA fragment geometry --------------------------------------------
  const int lane = t & 31;
  const int rl   = lane & 15;      // row-within-tile lane
  const int hid  = lane >> 4;      // K-half selector
  const int wave = t >> 5;         // 8 waves: 2 (M) x 4 (N)
  const int wm   = (wave >> 2) * 64;
  const int wn   = (wave & 3) * 64;

  v8f acc[4][4] = {};

  auto compute = [&](const _Float16* stage) {   // 16 WMMAs per k-step
    const _Float16* As = stage;
    const _Float16* Bs = stage + ATILE_H;
    Frag a[4], b[4];
#pragma unroll
    for (int mi = 0; mi < 4; ++mi) {
      const uint4* p = (const uint4*)(As + (wm + mi * 16 + rl) * LDA_H + hid * 8);
      a[mi].q[0] = p[0];        // K = hid*8 .. +7
      a[mi].q[1] = p[2];        // K = 16 + hid*8 .. +7
    }
#pragma unroll
    for (int ni = 0; ni < 4; ++ni) {
      const uint4* p = (const uint4*)(Bs + (wn + ni * 16 + rl) * LDA_H + hid * 8);
      b[ni].q[0] = p[0];
      b[ni].q[1] = p[2];
    }
#pragma unroll
    for (int mi = 0; mi < 4; ++mi)
#pragma unroll
      for (int ni = 0; ni < 4; ++ni)
        acc[mi][ni] = __builtin_amdgcn_wmma_f32_16x16x32_f16(
            false, a[mi].h, false, b[ni].h, (short)0, acc[mi][ni], false, false);
  };

#define SOFF(s) (ldsBase + (unsigned)((s) * (STAGE_H * 2)))

  // ---- 4-stage circular pipeline (fixed stage pointers per body) ----------
  issue(SOFF(0), 0);
  issue(SOFF(1), 1);
  issue(SOFF(2), 2);

  for (int kc = 0; kc < NKC; kc += 4) {
    wait_async_12();                       // stage kc landed (own loads)
    __syncthreads();                       // ... for every wave in the block
    compute(lds + 0 * STAGE_H);
    issue(SOFF(3), kc + 3);                // kc+3 <= 127: always valid

    wait_async_12();
    __syncthreads();
    compute(lds + 1 * STAGE_H);
    if (kc + 4 < NKC) issue(SOFF(0), kc + 4);

    wait_async_12();
    __syncthreads();
    compute(lds + 2 * STAGE_H);
    if (kc + 5 < NKC) issue(SOFF(1), kc + 5);

    wait_async_12();
    __syncthreads();
    compute(lds + 3 * STAGE_H);
    if (kc + 6 < NKC) issue(SOFF(2), kc + 6);
  }

  // ---- epilogue: per-channel scale + bias in fp32 -------------------------
  // C/D layout: lane<16 -> rows m0..m0+7, lane>=16 -> rows m0+8..m0+15; col = rl.
#pragma unroll
  for (int ni = 0; ni < 4; ++ni) {
    const int   o  = nBase + wn + ni * 16 + rl;
    const float sc = w_scale[o];
    const float bs = bias[o];
#pragma unroll
    for (int mi = 0; mi < 4; ++mi) {
      const int mrow = mBase + wm + mi * 16 + hid * 8;
#pragma unroll
      for (int v = 0; v < 8; ++v)
        out[(size_t)(mrow + v) * N_TOT + o] = acc[mi][ni][v] * sc + bs;
    }
  }
}

// ---------------------------------------------------------------------------
extern "C" void kernel_launch(void* const* d_in, const int* in_sizes, int n_in,
                              void* d_out, int out_size, void* d_ws, size_t ws_size,
                              hipStream_t stream) {
  const float* input   = (const float*)d_in[0];
  const int*   qweight = (const int*)d_in[1];
  const float* w_scale = (const float*)d_in[2];
  const int*   w_zp    = (const int*)d_in[3];
  const float* bias    = (const float*)d_in[4];
  float*       out     = (float*)d_out;

  _Float16* Ah = (_Float16*)d_ws;                                      // 67.1 MB
  _Float16* Wh = (_Float16*)((char*)d_ws + (size_t)M_TOT * K_TOT * 2); // 90.2 MB

  const unsigned gi = (unsigned)(((size_t)M_TOT * K_TOT) / (8 * 256)); // 16384
  const unsigned gw = (unsigned)(((size_t)N_TOT * K_TOT) / (8 * 256)); // 22016
  k_cvt_input<<<gi, 256, 0, stream>>>(input, Ah);
  k_cvt_weight<<<gw, 256, 0, stream>>>(qweight, w_zp, Wh);

  dim3 grid(N_TOT / BN, M_TOT / BM);   // 43 x 64
  k_qgemm<<<grid, 256, 0, stream>>>(Ah, Wh, w_scale, bias, out);
}
